// WindowAttention_29738353557857
// MI455X (gfx1250) — matrix operand order repaired and checked by reading, not compiled, not run on hardware
//
#include <hip/hip_runtime.h>
#include <hip/hip_bf16.h>

typedef __attribute__((ext_vector_type(16))) _Float16 v16h;
typedef __attribute__((ext_vector_type(8)))  _Float16 v8h;
typedef __attribute__((ext_vector_type(4)))  _Float16 v4h;
typedef __attribute__((ext_vector_type(8)))  float    v8f;
typedef __attribute__((ext_vector_type(4)))  float    v4f;

#define NWIN   49      // tokens per window (7x7)
#define NP     64      // padded tokens
#define CDIM   128     // channels
#define HEADS  4
#define HD     32      // head dim
#define XS     136     // LDS stride for [64][128] f16 tiles (+8 pad, 16B-aligned rows)
#define QS     40      // LDS stride for [64][32] q/k tiles
#define VTS    72      // LDS stride for transposed V [32][64] and P [64][64]
#define PS     72
#define SCALE  0.17677669529663687f  // 1/sqrt(32)

// ---- WMMA helper: D = A(16x32 f16) * B(32x16 f16) + C(f32) ----
__device__ __forceinline__ v8f wmma16(v16h a, v16h b, v8f c) {
  return __builtin_amdgcn_wmma_f32_16x16x32_f16(false, a, false, b, (short)0, c,
                                                false, false);
}

// A-operand (16x32, f16) from an LDS row-major tile.
// Layout: lanes 0-15 row=lane, elems 0-15 = K+0..15;
// lanes 16-31 same rows, K offset +16.
__device__ __forceinline__ v16h ldA(const _Float16* base, int stride,
                                    int rowbase, int kbase, int lane) {
  int r  = rowbase + (lane & 15);
  int k0 = kbase + ((lane >> 4) << 4);
  const _Float16* p = base + r * stride + k0;
  v8h lo = *(const v8h*)p;
  v8h hi = *(const v8h*)(p + 8);
  return __builtin_shufflevector(lo, hi, 0,1,2,3,4,5,6,7,8,9,10,11,12,13,14,15);
}

// B-operand (32x16, f16) where B[k][n] = rows[n][k] (i.e. B^T stored row-major
// in LDS). Contiguous k order used here (as in ldA): lanes 0-15 col=lane hold K=0..15, lanes 16-31 K=16..31.
__device__ __forceinline__ v16h ldBrows(const _Float16* base, int stride,
                                        int rowbase, int kbase, int lane) {
  int r  = rowbase + (lane & 15);
  int k0 = kbase + ((lane >> 4) << 4);
  const _Float16* p = base + r * stride + k0;
  v8h lo = *(const v8h*)p;
  v8h hi = *(const v8h*)(p + 8);
  return __builtin_shufflevector(lo, hi, 0,1,2,3,4,5,6,7,8,9,10,11,12,13,14,15);
}

// B-operand built from global f32 weight matrix W[rows][ld]:
// B[k][n] = W[rowbase+n][kbase+k]. 16 contiguous floats per lane -> cvt f16.
__device__ __forceinline__ v16h ldW(const float* __restrict__ W, int ld,
                                    int rowbase, int kbase, int lane) {
  int r  = rowbase + (lane & 15);
  int k0 = kbase + ((lane >> 4) << 4);
  const float* p = W + r * ld + k0;
  v4f f0 = *(const v4f*)(p);
  v4f f1 = *(const v4f*)(p + 4);
  v4f f2 = *(const v4f*)(p + 8);
  v4f f3 = *(const v4f*)(p + 12);
  v16h b;
  b[0]=(_Float16)f0.x; b[1]=(_Float16)f0.y; b[2]=(_Float16)f0.z; b[3]=(_Float16)f0.w;
  b[4]=(_Float16)f1.x; b[5]=(_Float16)f1.y; b[6]=(_Float16)f1.z; b[7]=(_Float16)f1.w;
  b[8]=(_Float16)f2.x; b[9]=(_Float16)f2.y; b[10]=(_Float16)f2.z; b[11]=(_Float16)f2.w;
  b[12]=(_Float16)f3.x; b[13]=(_Float16)f3.y; b[14]=(_Float16)f3.z; b[15]=(_Float16)f3.w;
  return b;
}

__global__ __launch_bounds__(128)
void swin_window_attn_kernel(const float* __restrict__ x,
                             const float* __restrict__ mask,
                             const float* __restrict__ qkv_w,
                             const float* __restrict__ qkv_b,
                             const float* __restrict__ proj_w,
                             const float* __restrict__ proj_b,
                             const float* __restrict__ bias_table,
                             const int*   __restrict__ rel_index,
                             float* __restrict__ out) {
  // LDS: sQ[4][64][40] | sK[4][64][40] | sVt[4][32][72] | union{ sX[64][136], sP[4][64][72], sO[64][136] }
  __shared__ __align__(16) _Float16 smem[10240 + 10240 + 9216 + 18432];
  _Float16* sQ  = smem;
  _Float16* sK  = smem + 10240;
  _Float16* sVt = smem + 20480;
  _Float16* sUni = smem + 29696;

  const int tid  = threadIdx.x;
  const int lane = tid & 31;
  const int wave = tid >> 5;
  const int b    = blockIdx.x;
  const int win  = b & 63;           // B_ = 128 * 64 windows -> mask index
  const int hi8  = (lane >> 4) << 3; // row offset within C/D tile
  const int cm   = lane & 15;        // column within tile

  // ---------- Phase 0: stage x (f32 -> f16) into sX, zero pad rows ----------
  _Float16* sX = sUni;
  const float* xg = x + (size_t)b * (NWIN * CDIM);
  for (int i = tid; i < (NWIN * CDIM) / 4; i += 128) {
    v4f f = *(const v4f*)(xg + i * 4);
    int flat = i * 4;
    int row = flat >> 7, col = flat & 127;
    v4h h4; h4[0]=(_Float16)f.x; h4[1]=(_Float16)f.y; h4[2]=(_Float16)f.z; h4[3]=(_Float16)f.w;
    *(v4h*)&sX[row * XS + col] = h4;
  }
  for (int i = tid; i < (NP - NWIN) * XS; i += 128)
    sX[NWIN * XS + i] = (_Float16)0.f;
  __syncthreads();

  // ---------- Phase 1: QKV GEMM [64x128] x [128x384] ----------
  // col tile t: s = t/8 (q/k/v), head = (t%8)/2, half = t%2
  for (int tt = 0; tt < 6; ++tt) {
    int t = wave * 6 + tt;
    int s = t >> 3, h2 = (t & 7) >> 1, dh = t & 1;
    int obase = t << 4;
    float qb = qkv_b[obase + cm];
    v8f acc[4] = {};
    for (int kk = 0; kk < CDIM; kk += 32) {
      v16h bt = ldW(qkv_w, CDIM, obase, kk, lane);
#pragma unroll
      for (int i = 0; i < 4; ++i) {
        v16h at = ldA(sX, XS, i * 16, kk, lane);
        acc[i] = wmma16(at, bt, acc[i]);
      }
    }
    if (s < 2) {  // Q (pre-scaled) / K, row-major [head][n][d]
      _Float16* dst = (s == 0 ? sQ : sK) + h2 * NP * QS + dh * 16 + cm;
      float scl = (s == 0) ? SCALE : 1.0f;
#pragma unroll
      for (int i = 0; i < 4; ++i)
#pragma unroll
        for (int r = 0; r < 8; ++r) {
          int row = i * 16 + hi8 + r;
          dst[row * QS] = (_Float16)((acc[i][r] + qb) * scl);
        }
    } else {      // V transposed: sVt[head][d][m] -> packed 16B stores
      int d = dh * 16 + cm;
      _Float16* dst = sVt + (h2 * HD + d) * VTS;
#pragma unroll
      for (int i = 0; i < 4; ++i) {
        v8h tv;
#pragma unroll
        for (int r = 0; r < 8; ++r) tv[r] = (_Float16)(acc[i][r] + qb);
        *(v8h*)&dst[i * 16 + hi8] = tv;
      }
    }
  }
  __syncthreads();

  // ---------- Phase 2: per-head attention (wave h owns head h) ----------
  const int h = wave;
  const _Float16* qh  = sQ  + h * NP * QS;
  const _Float16* kh  = sK  + h * NP * QS;
  const _Float16* vth = sVt + h * HD * VTS;
  _Float16* ph = sUni + h * NP * PS;   // P (softmax probs, f16), overlays dead sX

  for (int i = 0; i < 4; ++i) {
    v16h qa = ldA(qh, QS, i * 16, 0, lane);   // K-dim = hd = 32: single step
    v8f sc[4];
#pragma unroll
    for (int j = 0; j < 4; ++j) {
      v16h kb = ldBrows(kh, QS, j * 16, 0, lane);
      v8f z = {};
      sc[j] = wmma16(qa, kb, z);
    }
    // bias gather + mask + padded-key masking + row softmax
#pragma unroll
    for (int r = 0; r < 8; ++r) {
      int n = i * 16 + hi8 + r;
      float v[4];
      float mx = -1e30f;
#pragma unroll
      for (int j = 0; j < 4; ++j) {
        int m = j * 16 + cm;
        float sv;
        if (n < NWIN && m < NWIN) {
          int ri = rel_index[n * NWIN + m];
          sv = sc[j][r] + bias_table[ri * HEADS + h]
             + mask[((size_t)win * NWIN + n) * NWIN + m];
        } else {
          sv = -1e30f;
        }
        v[j] = sv;
        mx = fmaxf(mx, sv);
      }
      mx = fmaxf(mx, __shfl_xor(mx, 1, 32));
      mx = fmaxf(mx, __shfl_xor(mx, 2, 32));
      mx = fmaxf(mx, __shfl_xor(mx, 4, 32));
      mx = fmaxf(mx, __shfl_xor(mx, 8, 32));
      float sum = 0.f;
#pragma unroll
      for (int j = 0; j < 4; ++j) { v[j] = __expf(v[j] - mx); sum += v[j]; }
      sum += __shfl_xor(sum, 1, 32);
      sum += __shfl_xor(sum, 2, 32);
      sum += __shfl_xor(sum, 4, 32);
      sum += __shfl_xor(sum, 8, 32);
      float inv = __builtin_amdgcn_rcpf(sum);
#pragma unroll
      for (int j = 0; j < 4; ++j)
        ph[n * PS + j * 16 + cm] = (_Float16)(v[j] * inv);
    }
  }
  __syncthreads();

  // ---------- Phase 3: out = P[64x64] @ V[64x32] ----------
  v8f o0[4] = {}, o1[4] = {};
  for (int kk = 0; kk < NP; kk += 32) {
    v16h vb0 = ldBrows(vth, VTS, 0, kk, lane);   // B[m][d] from sVt rows d
    v16h vb1 = ldBrows(vth, VTS, 16, kk, lane);
#pragma unroll
    for (int i = 0; i < 4; ++i) {
      v16h pa = ldA(ph, PS, i * 16, kk, lane);
      o0[i] = wmma16(pa, vb0, o0[i]);
      o1[i] = wmma16(pa, vb1, o1[i]);
    }
  }
  __syncthreads();   // all P reads done before sUni is reused as sO

  _Float16* sO = sUni;  // [64][136], col c = h*32 + d
#pragma unroll
  for (int i = 0; i < 4; ++i)
#pragma unroll
    for (int r = 0; r < 8; ++r) {
      int n = i * 16 + hi8 + r;
      sO[n * XS + h * HD + cm]      = (_Float16)o0[i][r];
      sO[n * XS + h * HD + 16 + cm] = (_Float16)o1[i][r];
    }
  __syncthreads();

  // ---------- Phase 4: proj GEMM [64x128] x [128x128] + bias, store n<49 ----------
#pragma unroll
  for (int nt = 0; nt < 2; ++nt) {
    int cb = (wave * 2 + nt) * 16;
    float pb = proj_b[cb + cm];
    v8f acc[4] = {};
    for (int kk = 0; kk < CDIM; kk += 32) {
      v16h bt = ldW(proj_w, CDIM, cb, kk, lane);
#pragma unroll
      for (int i = 0; i < 4; ++i) {
        v16h at = ldA(sO, XS, i * 16, kk, lane);
        acc[i] = wmma16(at, bt, acc[i]);
      }
    }
    float* og = out + (size_t)b * (NWIN * CDIM) + cb + cm;
#pragma unroll
    for (int i = 0; i < 4; ++i)
#pragma unroll
      for (int r = 0; r < 8; ++r) {
        int n = i * 16 + hi8 + r;
        if (n < NWIN) og[(size_t)n * CDIM] = acc[i][r] + pb;
      }
  }
}

extern "C" void kernel_launch(void* const* d_in, const int* in_sizes, int n_in,
                              void* d_out, int out_size, void* d_ws, size_t ws_size,
                              hipStream_t stream) {
  (void)n_in; (void)out_size; (void)d_ws; (void)ws_size;
  const float* x          = (const float*)d_in[0];
  const float* mask       = (const float*)d_in[1];
  const float* qkv_w      = (const float*)d_in[2];
  const float* qkv_b      = (const float*)d_in[3];
  const float* proj_w     = (const float*)d_in[4];
  const float* proj_b     = (const float*)d_in[5];
  const float* bias_table = (const float*)d_in[6];
  const int*   rel_index  = (const int*)d_in[7];
  float* outp = (float*)d_out;

  int B_ = in_sizes[0] / (NWIN * CDIM);  // 8192 windows
  swin_window_attn_kernel<<<B_, 128, 0, stream>>>(
      x, mask, qkv_w, qkv_b, proj_w, proj_b, bias_table, rel_index, outp);
}
